// BasicAggModel_59863254172515
// MI455X (gfx1250) — matrix operand order repaired and checked by reading, not compiled, not run on hardware
//
#include <hip/hip_runtime.h>
#include <hip/hip_bf16.h>

// ---------------------------------------------------------------------------
// Model dims
// ---------------------------------------------------------------------------
#define Bsz   2048
#define Nn    64
#define Dd    256
#define Hh    8
#define DInn  2048
#define NSv   1298
#define NOv   38

typedef __bf16 bf16_t;
typedef __bf16  v16bf __attribute__((ext_vector_type(16)));
typedef __bf16  bf8v  __attribute__((ext_vector_type(8)));
typedef __bf16  bf4v  __attribute__((ext_vector_type(4)));
typedef float   v8f   __attribute__((ext_vector_type(8)));

// async copy: global -> LDS, 16 bytes per lane, tracked by ASYNCcnt
#define ASYNC_B128(ldsoff, gptr)                                              \
    asm volatile("global_load_async_to_lds_b128 %0, %1, off"                  \
                 :: "v"(ldsoff), "v"(gptr) : "memory")
#define WAIT_ASYNC() asm volatile("s_wait_asynccnt 0x0" ::: "memory")

// ---------------------------------------------------------------------------
// f32 -> bf16 convert (4-wide; n must be a multiple of 4)
// ---------------------------------------------------------------------------
__global__ void f2bf_kernel(const float* __restrict__ X, bf16_t* __restrict__ Y, int n) {
    int i = blockIdx.x * blockDim.x + threadIdx.x;
    int base = i * 4;
    if (base < n) {
        float4 x = *(const float4*)&X[base];
        bf4v y;
        y[0] = (bf16_t)x.x; y[1] = (bf16_t)x.y; y[2] = (bf16_t)x.z; y[3] = (bf16_t)x.w;
        *(bf4v*)&Y[base] = y;
    }
}

// ---------------------------------------------------------------------------
// W_comb[h*256+c, out] = sum_d w_vs[h*64+d, c] * fc_w[out, h*64+d]
// ---------------------------------------------------------------------------
__global__ __launch_bounds__(256)
void wcomb_kernel(const float* __restrict__ w_vs, const float* __restrict__ fc_w,
                  bf16_t* __restrict__ W) {
    int gid = blockIdx.x * 256 + threadIdx.x;
    int out = gid & 255;
    int j   = gid >> 8;            // 0..2047  (= h*256 + c)
    int h   = j >> 8;
    int c   = j & 255;
    float acc = 0.f;
    #pragma unroll 8
    for (int d = 0; d < 64; ++d)
        acc += w_vs[(h * 64 + d) * 256 + c] * fc_w[out * 512 + h * 64 + d];
    W[(size_t)j * 256 + out] = (bf16_t)acc;
}

// ---------------------------------------------------------------------------
// Generic LDS-staged bf16 WMMA GEMM:  C[M,N] = A[M,K] * B
//   A  : row-major bf16, lda
//   Bm : btrans==0 -> B[k*ldb + n] (KxN) ; btrans==1 -> B[n*ldb + k] (NxK, x@W^T)
//   C  : optional f32 out ; Cbf : optional bf16 out (same ldc)
// Block: 256 thr = 8 waves; tile 64x64, K-step 32; wave tile 16x32 -> 2 WMMA.
// B tile stored transposed in LDS (Bt[n][k]) so fragments are contiguous b128.
// Fast path stages via global_load_async_to_lds_b128 (ASYNCcnt).
// K must be a multiple of 32.
// ---------------------------------------------------------------------------
#define TM 64
#define TN 64
#define TK 32
#define LPAD 8   // +16B row pad, keeps 16B alignment (row stride 80B)

__global__ __launch_bounds__(256)
void gemm_bf16_kernel(const bf16_t* __restrict__ A, int lda,
                      const bf16_t* __restrict__ Bm, int ldb, int btrans,
                      float* __restrict__ C, int ldc,
                      bf16_t* __restrict__ Cbf,
                      int M, int N, int K) {
    __shared__ bf16_t As[TM][TK + LPAD];
    __shared__ bf16_t Bt[TN][TK + LPAD];

    const int tid  = threadIdx.x;
    const int lane = tid & 31;
    const int wid  = tid >> 5;
    const int wm   = wid & 3;        // 4 waves along M
    const int wn   = wid >> 2;       // 2 waves along N
    const int m0   = blockIdx.x * TM;
    const int n0   = blockIdx.y * TN;

    const bool edge = (m0 + TM > M) || (n0 + TN > N);

    v8f acc0, acc1;
    #pragma unroll
    for (int r = 0; r < 8; ++r) { acc0[r] = 0.f; acc1[r] = 0.f; }

    // fast-path staging coordinates (16B chunks)
    const int ar = tid >> 2;            // A row 0..63
    const int ac = (tid & 3) * 8;       // A col {0,8,16,24}
    const int bnN = tid >> 2;           // NT: B row (n) 0..63
    const int bkN = (tid & 3) * 8;      //     k {0,8,16,24}
    const int bnG = tid & 63;           // NN: n 0..63
    const int bkG = (tid >> 6) * 8;     //     k {0,8,16,24}

    const unsigned lds_a  = (unsigned)(size_t)&As[ar][ac];
    const unsigned lds_bt = (unsigned)(size_t)&Bt[bnN][bkN];

    // hoisted, strength-reduced global pointers (stepped per K-iter)
    const bf16_t* pa  = A  + (size_t)(m0 + ar)  * lda + ac;                 // async A
    const bf16_t* pbt = Bm + (size_t)(n0 + bnN) * ldb + bkN;                // NT async B
    const bf16_t* pbg = Bm + (size_t)bkG * ldb + (n0 + bnG);                // NN gather B
    const size_t  ldbS   = (size_t)ldb;
    const size_t  stepBg = (size_t)TK * ldb;

    for (int k0 = 0; k0 < K; k0 += TK) {
        if (!edge) {
            // A tile: async b128 per lane
            ASYNC_B128(lds_a, pa);
            if (btrans) {
                // NT: rows of W are K-contiguous -> async b128 straight into Bt
                ASYNC_B128(lds_bt, pbt);
                pbt += TK;
            } else {
                // NN: K-strided gather, pointer-stepped, one ds_store_b128
                const bf16_t* q = pbg;
                bf8v pk;
                #pragma unroll
                for (int i = 0; i < 8; ++i) { pk[i] = *q; q += ldbS; }
                *(bf8v*)&Bt[bnG][bkG] = pk;
                pbg += stepBg;
            }
            pa += TK;
            WAIT_ASYNC();
        } else {
            #pragma unroll
            for (int i = 0; i < 8; ++i) {
                int gm = m0 + ar, gk = k0 + ac + i;
                As[ar][ac + i] = (gm < M) ? A[(size_t)gm * lda + gk] : (bf16_t)0.f;
            }
            const int gn = n0 + bnG;
            if (gn < N) {
                bf8v pk;
                if (btrans) {
                    const bf16_t* q = Bm + (size_t)gn * ldb + k0 + bkG;
                    #pragma unroll
                    for (int i = 0; i < 8; ++i) pk[i] = q[i];
                } else {
                    const bf16_t* q = Bm + (size_t)(k0 + bkG) * ldb + gn;
                    #pragma unroll
                    for (int i = 0; i < 8; ++i) { pk[i] = *q; q += ldbS; }
                }
                *(bf8v*)&Bt[bnG][bkG] = pk;
            } else {
                bf8v pk;
                #pragma unroll
                for (int i = 0; i < 8; ++i) pk[i] = (bf16_t)0.f;
                *(bf8v*)&Bt[bnG][bkG] = pk;
            }
        }
        __syncthreads();

        // fragments: contiguous k order (lane half h holds k=16h..16h+15), all b128 LDS reads
        const int am = wm * 16 + (lane & 15);
        const int kb = (lane < 16) ? 0 : 16;    // A: K=0..15 vs 16..31
        bf8v alo = *(const bf8v*)&As[am][kb];
        bf8v ahi = *(const bf8v*)&As[am][kb + 8];
        v16bf afrag = __builtin_shufflevector(alo, ahi,
                        0,1,2,3,4,5,6,7,8,9,10,11,12,13,14,15);

        const int bn  = wn * 32 + (lane & 15);
        const int kbb = (lane < 16) ? 0 : 16;   // B: K=0..15 vs 16..31
        bf8v b0lo = *(const bf8v*)&Bt[bn][kbb];
        bf8v b0hi = *(const bf8v*)&Bt[bn][kbb + 8];
        v16bf b0 = __builtin_shufflevector(b0lo, b0hi,
                        0,1,2,3,4,5,6,7,8,9,10,11,12,13,14,15);
        bf8v b1lo = *(const bf8v*)&Bt[bn + 16][kbb];
        bf8v b1hi = *(const bf8v*)&Bt[bn + 16][kbb + 8];
        v16bf b1 = __builtin_shufflevector(b1lo, b1hi,
                        0,1,2,3,4,5,6,7,8,9,10,11,12,13,14,15);

        acc0 = __builtin_amdgcn_wmma_f32_16x16x32_bf16(false, afrag, false, b0,
                                                       (short)0, acc0, false, false);
        acc1 = __builtin_amdgcn_wmma_f32_16x16x32_bf16(false, afrag, false, b1,
                                                       (short)0, acc1, false, false);
        __syncthreads();
    }

    // store C (CDNA5 16x16 f32 C/D layout)
    const int cn0 = n0 + wn * 32 + (lane & 15);
    #pragma unroll
    for (int r = 0; r < 8; ++r) {
        int cm = m0 + wm * 16 + ((lane < 16) ? r : 8 + r);
        if (cm < M) {
            if (cn0 < N) {
                if (C)   C  [(size_t)cm * ldc + cn0] = acc0[r];
                if (Cbf) Cbf[(size_t)cm * ldc + cn0] = (bf16_t)acc0[r];
            }
            int cn1 = cn0 + 16;
            if (cn1 < N) {
                if (C)   C  [(size_t)cm * ldc + cn1] = acc1[r];
                if (Cbf) Cbf[(size_t)cm * ldc + cn1] = (bf16_t)acc1[r];
            }
        }
    }
}

// ---------------------------------------------------------------------------
// Fused attention (grid.x = B, 256 thr = 8 waves):
//  x = LN(PE_states+POS) on the fly; logits = 0.5*(x.qkproj/8 + score), masked;
//  shared softmax; agg_{s,o}[b,h,:] = sum_n p[h,n]*emb rows (LDS-staged bf16).
// ---------------------------------------------------------------------------
__global__ __launch_bounds__(256)
void attn_kernel(const float* __restrict__ pe_states,
                 const bf16_t* __restrict__ qkproj,      // [B, H*256]
                 const float* __restrict__ scores,       // [B, N]
                 const float* __restrict__ stmt_emb,     // [NS+1, 256]
                 const float* __restrict__ op_emb,       // [NO+1, 256]
                 const int*   __restrict__ stmts,
                 const int*   __restrict__ ops,
                 const float* __restrict__ lnw,
                 const float* __restrict__ lnb,
                 bf16_t* __restrict__ aggs,              // [B, H*256] bf16
                 bf16_t* __restrict__ aggo) {
    __shared__ bf16_t vlds[Nn][Dd + 8];    // row stride 528B = 33*16 (aligned)
    __shared__ float  prob[Hh][Nn];
    __shared__ float  sc[Nn];
    __shared__ int    st[Nn];
    __shared__ int    opx[Nn];

    const int b = blockIdx.x;
    const int tid = threadIdx.x, lane = tid & 31, wid = tid >> 5;

    if (tid < Nn) {
        st [tid] = stmts [b * Nn + tid];
        opx[tid] = ops   [b * Nn + tid];
        sc [tid] = scores[b * Nn + tid];
    }
    __syncthreads();

    const float l2t = 13.287712379549449f;   // log2(10000)

    for (int i = 0; i < 8; ++i) {
        const int n = wid * 8 + i;
        float xh[8];
        const float4 xa = *(const float4*)&pe_states[((size_t)b * Nn + n) * Dd + lane * 8];
        const float4 xb = *(const float4*)&pe_states[((size_t)b * Nn + n) * Dd + lane * 8 + 4];
        xh[0] = xa.x; xh[1] = xa.y; xh[2] = xa.z; xh[3] = xa.w;
        xh[4] = xb.x; xh[5] = xb.y; xh[6] = xb.z; xh[7] = xb.w;
        float s = 0.f, ss = 0.f;
        #pragma unroll
        for (int j = 0; j < 8; ++j) {
            int c = lane * 8 + j;
            float ang = (float)n * exp2f(-l2t * ((float)((c >> 1) << 1) / (float)Dd));
            float pos = (c & 1) ? cosf(ang) : sinf(ang);
            float x = xh[j] + pos;
            xh[j] = x; s += x; ss += x * x;
        }
        #pragma unroll
        for (int off = 16; off > 0; off >>= 1) {
            s  += __shfl_xor(s,  off, 32);
            ss += __shfl_xor(ss, off, 32);
        }
        float mean = s * (1.f / Dd);
        float var  = ss * (1.f / Dd) - mean * mean;
        float rstd = rsqrtf(var + 1e-6f);
        #pragma unroll
        for (int j = 0; j < 8; ++j) {
            int c = lane * 8 + j;
            xh[j] = (xh[j] - mean) * rstd * lnw[c] + lnb[c];
        }
        // gather statement value row -> LDS (one b128 store)
        {
            int r = st[n];
            const float4 ea = *(const float4*)&stmt_emb[(size_t)r * Dd + lane * 8];
            const float4 eb = *(const float4*)&stmt_emb[(size_t)r * Dd + lane * 8 + 4];
            bf8v pk;
            pk[0] = (bf16_t)ea.x; pk[1] = (bf16_t)ea.y; pk[2] = (bf16_t)ea.z; pk[3] = (bf16_t)ea.w;
            pk[4] = (bf16_t)eb.x; pk[5] = (bf16_t)eb.y; pk[6] = (bf16_t)eb.z; pk[7] = (bf16_t)eb.w;
            *(bf8v*)&vlds[n][lane * 8] = pk;
        }
        // logits for all 8 heads (16B qkproj loads)
        const bool valid = (st[n] != NSv);
        #pragma unroll
        for (int h = 0; h < Hh; ++h) {
            bf8v qv = *(const bf8v*)&qkproj[((size_t)b * Hh + h) * Dd + lane * 8];
            float d = 0.f;
            #pragma unroll
            for (int j = 0; j < 8; ++j) d += xh[j] * (float)qv[j];
            #pragma unroll
            for (int off = 16; off > 0; off >>= 1) d += __shfl_xor(d, off, 32);
            if (lane == 0)
                prob[h][n] = valid ? 0.5f * (d * 0.125f + sc[n]) : -1e9f;
        }
    }
    __syncthreads();

    // masked softmax over n: wave wid owns head wid
    {
        const int h = wid;
        float v0 = prob[h][lane], v1 = prob[h][lane + 32];
        float m = fmaxf(v0, v1);
        #pragma unroll
        for (int off = 16; off > 0; off >>= 1) m = fmaxf(m, __shfl_xor(m, off, 32));
        float e0 = expf(v0 - m), e1 = expf(v1 - m);
        float s = e0 + e1;
        #pragma unroll
        for (int off = 16; off > 0; off >>= 1) s += __shfl_xor(s, off, 32);
        float inv = 1.f / s;
        prob[h][lane]      = e0 * inv;
        prob[h][lane + 32] = e1 * inv;
    }
    __syncthreads();

    // aggregate statement values (b128 LDS reads, b128 global store)
    {
        const int h = wid;
        float acc[8];
        #pragma unroll
        for (int j = 0; j < 8; ++j) acc[j] = 0.f;
        for (int n = 0; n < Nn; ++n) {
            float p = prob[h][n];
            bf8v vv = *(const bf8v*)&vlds[n][lane * 8];
            #pragma unroll
            for (int j = 0; j < 8; ++j) acc[j] += p * (float)vv[j];
        }
        bf8v ov;
        #pragma unroll
        for (int j = 0; j < 8; ++j) ov[j] = (bf16_t)acc[j];
        *(bf8v*)&aggs[((size_t)b * Hh + h) * Dd + lane * 8] = ov;
    }
    __syncthreads();

    // reload LDS with operator value rows
    for (int i = 0; i < 8; ++i) {
        const int n = wid * 8 + i;
        int r = opx[n];
        const float4 ea = *(const float4*)&op_emb[(size_t)r * Dd + lane * 8];
        const float4 eb = *(const float4*)&op_emb[(size_t)r * Dd + lane * 8 + 4];
        bf8v pk;
        pk[0] = (bf16_t)ea.x; pk[1] = (bf16_t)ea.y; pk[2] = (bf16_t)ea.z; pk[3] = (bf16_t)ea.w;
        pk[4] = (bf16_t)eb.x; pk[5] = (bf16_t)eb.y; pk[6] = (bf16_t)eb.z; pk[7] = (bf16_t)eb.w;
        *(bf8v*)&vlds[n][lane * 8] = pk;
    }
    __syncthreads();

    {
        const int h = wid;
        float acc[8];
        #pragma unroll
        for (int j = 0; j < 8; ++j) acc[j] = 0.f;
        for (int n = 0; n < Nn; ++n) {
            float p = prob[h][n];
            bf8v vv = *(const bf8v*)&vlds[n][lane * 8];
            #pragma unroll
            for (int j = 0; j < 8; ++j) acc[j] += p * (float)vv[j];
        }
        bf8v ov;
        #pragma unroll
        for (int j = 0; j < 8; ++j) ov[j] = (bf16_t)acc[j];
        *(bf8v*)&aggo[((size_t)b * Hh + h) * Dd + lane * 8] = ov;
    }
}

// ---------------------------------------------------------------------------
// Row LayerNorm: y = LN(X + res? + cbias?) * gamma + beta ; one row per block
// ---------------------------------------------------------------------------
__global__ __launch_bounds__(256)
void ln_rows_kernel(const float* __restrict__ X, const float* __restrict__ res,
                    const float* __restrict__ cbias,
                    const float* __restrict__ gamma, const float* __restrict__ beta,
                    float* __restrict__ Yf, bf16_t* __restrict__ Ybf, int cols) {
    __shared__ float red[16];
    const int row = blockIdx.x, tid = threadIdx.x, lane = tid & 31, wid = tid >> 5;
    float v[8];
    float s = 0.f, ss = 0.f;
    for (int c = tid, idx = 0; c < cols; c += 256, ++idx) {
        float x = X[(size_t)row * cols + c];
        if (res)   x += res[(size_t)row * cols + c];
        if (cbias) x += cbias[c];
        v[idx] = x; s += x; ss += x * x;
    }
    #pragma unroll
    for (int off = 16; off > 0; off >>= 1) {
        s  += __shfl_xor(s,  off, 32);
        ss += __shfl_xor(ss, off, 32);
    }
    if (lane == 0) { red[wid] = s; red[8 + wid] = ss; }
    __syncthreads();
    if (wid == 0) {
        float a  = (lane < 8) ? red[lane]     : 0.f;
        float b2 = (lane < 8) ? red[8 + lane] : 0.f;
        #pragma unroll
        for (int off = 4; off > 0; off >>= 1) {
            a  += __shfl_xor(a,  off, 32);
            b2 += __shfl_xor(b2, off, 32);
        }
        if (lane == 0) { red[0] = a; red[1] = b2; }
    }
    __syncthreads();
    const float mean = red[0] / (float)cols;
    const float var  = red[1] / (float)cols - mean * mean;
    const float rstd = rsqrtf(var + 1e-6f);
    for (int c = tid, idx = 0; c < cols; c += 256, ++idx) {
        float y = (v[idx] - mean) * rstd * gamma[c] + beta[c];
        if (Yf)  Yf [(size_t)row * cols + c] = y;
        if (Ybf) Ybf[(size_t)row * cols + c] = (bf16_t)y;
    }
}

// ---------------------------------------------------------------------------
// relu(X + bias[col]) -> bf16, 4-wide (cols power of 2)
// ---------------------------------------------------------------------------
__global__ void relu_bias_kernel(const float* __restrict__ X, const float* __restrict__ bias,
                                 bf16_t* __restrict__ Y, int total, int cols) {
    int base = (blockIdx.x * blockDim.x + threadIdx.x) * 4;
    if (base < total) {
        float4 x = *(const float4*)&X[base];
        const float4 bb = *(const float4*)&bias[base & (cols - 1)];
        bf4v y;
        y[0] = (bf16_t)fmaxf(x.x + bb.x, 0.f);
        y[1] = (bf16_t)fmaxf(x.y + bb.y, 0.f);
        y[2] = (bf16_t)fmaxf(x.z + bb.z, 0.f);
        y[3] = (bf16_t)fmaxf(x.w + bb.w, 0.f);
        *(bf4v*)&Y[base] = y;
    }
}

// ---------------------------------------------------------------------------
// launch
// ---------------------------------------------------------------------------
extern "C" void kernel_launch(void* const* d_in, const int* in_sizes, int n_in,
                              void* d_out, int out_size, void* d_ws, size_t ws_size,
                              hipStream_t stream) {
    (void)in_sizes; (void)n_in; (void)out_size; (void)ws_size;

    const float* pe_states = (const float*)d_in[0];
    const float* gstate    = (const float*)d_in[1];
    const float* scores    = (const float*)d_in[2];
    const float* stmt_emb  = (const float*)d_in[3];
    const float* op_emb    = (const float*)d_in[4];
    const float* w_qs      = (const float*)d_in[5];
    const float* w_ks      = (const float*)d_in[6];
    const float* w_vs      = (const float*)d_in[7];
    const float* fc_w      = (const float*)d_in[8];
    const float* mha_ln_w  = (const float*)d_in[9];
    const float* mha_ln_b  = (const float*)d_in[10];
    const float* mdl_ln_w  = (const float*)d_in[11];
    const float* mdl_ln_b  = (const float*)d_in[12];
    const float* s_ff_w1   = (const float*)d_in[13];
    const float* s_ff_b1   = (const float*)d_in[14];
    const float* s_ff_w2   = (const float*)d_in[15];
    const float* s_ff_b2   = (const float*)d_in[16];
    const float* s_ff_lnw  = (const float*)d_in[17];
    const float* s_ff_lnb  = (const float*)d_in[18];
    const float* o_ff_w1   = (const float*)d_in[19];
    const float* o_ff_b1   = (const float*)d_in[20];
    const float* o_ff_w2   = (const float*)d_in[21];
    const float* o_ff_b2   = (const float*)d_in[22];
    const float* o_ff_lnw  = (const float*)d_in[23];
    const float* o_ff_lnb  = (const float*)d_in[24];
    const float* s_proj    = (const float*)d_in[25];
    const float* o_proj    = (const float*)d_in[26];
    const float* s_out_w   = (const float*)d_in[27];
    const float* s_out_b   = (const float*)d_in[28];
    const float* o_out_w   = (const float*)d_in[29];
    const float* o_out_b   = (const float*)d_in[30];
    const int*   stmts     = (const int*)d_in[31];
    const int*   ops       = (const int*)d_in[32];

    float* out_s = (float*)d_out;
    float* out_o = out_s + (size_t)Bsz * NSv;

    // workspace carve-up
    char* base = (char*)d_ws;
    size_t cur = 0;
    auto alloc = [&](size_t bytes) -> void* {
        cur = (cur + 255) & ~(size_t)255;
        void* p = base + cur;
        cur += bytes;
        return p;
    };
    bf16_t* gs_bf     = (bf16_t*)alloc((size_t)Bsz * Dd * 2);
    bf16_t* wqs_bf    = (bf16_t*)alloc((size_t)512 * 256 * 2);
    bf16_t* wks_bf    = (bf16_t*)alloc((size_t)512 * 256 * 2);
    bf16_t* sffw1_bf  = (bf16_t*)alloc((size_t)DInn * Dd * 2);
    bf16_t* sffw2_bf  = (bf16_t*)alloc((size_t)Dd * DInn * 2);
    bf16_t* offw1_bf  = (bf16_t*)alloc((size_t)DInn * Dd * 2);
    bf16_t* offw2_bf  = (bf16_t*)alloc((size_t)Dd * DInn * 2);
    bf16_t* sproj_bf  = (bf16_t*)alloc((size_t)NSv * Dd * 2);
    bf16_t* oproj_bf  = (bf16_t*)alloc((size_t)NOv * Dd * 2);
    bf16_t* wcomb_bf  = (bf16_t*)alloc((size_t)2048 * 256 * 2);
    bf16_t* qh_bf     = (bf16_t*)alloc((size_t)Bsz * 512 * 2);
    bf16_t* qkproj_bf = (bf16_t*)alloc((size_t)Bsz * 2048 * 2);
    bf16_t* aggs_bf   = (bf16_t*)alloc((size_t)Bsz * 2048 * 2);
    bf16_t* aggo_bf   = (bf16_t*)alloc((size_t)Bsz * 2048 * 2);
    float*  y_f       = (float*)alloc((size_t)Bsz * Dd * 4);
    float*  x1_f      = (float*)alloc((size_t)Bsz * Dd * 4);
    bf16_t* x1_bf     = (bf16_t*)alloc((size_t)Bsz * Dd * 2);
    float*  t_f       = (float*)alloc((size_t)Bsz * DInn * 4);
    bf16_t* h1_bf     = (bf16_t*)alloc((size_t)Bsz * DInn * 2);
    float*  u_f       = (float*)alloc((size_t)Bsz * Dd * 4);
    bf16_t* x2_bf     = (bf16_t*)alloc((size_t)Bsz * Dd * 2);
    float*  lgt_f     = (float*)alloc((size_t)Bsz * NSv * 4);

    auto cvt = [&](const float* src, bf16_t* dst, int n) {
        f2bf_kernel<<<(n / 4 + 255) / 256, 256, 0, stream>>>(src, dst, n);
    };

    // weight / activation converts
    cvt(gstate,   gs_bf,    Bsz * Dd);
    cvt(w_qs,     wqs_bf,   512 * 256);
    cvt(w_ks,     wks_bf,   512 * 256);
    cvt(s_ff_w1,  sffw1_bf, DInn * Dd);
    cvt(s_ff_w2,  sffw2_bf, Dd * DInn);
    cvt(o_ff_w1,  offw1_bf, DInn * Dd);
    cvt(o_ff_w2,  offw2_bf, Dd * DInn);
    cvt(s_proj,   sproj_bf, NSv * Dd);
    cvt(o_proj,   oproj_bf, NOv * Dd);

    // fused (w_vs, fc_w) weight
    wcomb_kernel<<<2048, 256, 0, stream>>>(w_vs, fc_w, wcomb_bf);

    // qh = global_state @ w_qs^T   [2048 x 512]
    gemm_bf16_kernel<<<dim3(32, 8), 256, 0, stream>>>(
        gs_bf, Dd, wqs_bf, Dd, 1, nullptr, 512, qh_bf, Bsz, 512, Dd);

    // qkproj[b, h*256+c] = qh[b, h*64+:64] @ w_ks_h  (8 head GEMMs, K=64)
    for (int h = 0; h < Hh; ++h) {
        gemm_bf16_kernel<<<dim3(32, 4), 256, 0, stream>>>(
            qh_bf + h * 64, 512,
            wks_bf + (size_t)h * 64 * 256, 256, 0,
            nullptr, 2048, qkproj_bf + h * 256,
            Bsz, 256, 64);
    }

    // fused LN + attention + dual value aggregation
    attn_kernel<<<Bsz, 256, 0, stream>>>(
        pe_states, qkproj_bf, scores, stmt_emb, op_emb, stmts, ops,
        mdl_ln_w, mdl_ln_b, aggs_bf, aggo_bf);

    // ---- statement stream ----
    gemm_bf16_kernel<<<dim3(32, 4), 256, 0, stream>>>(
        aggs_bf, 2048, wcomb_bf, 256, 0, y_f, Dd, nullptr, Bsz, Dd, 2048);
    ln_rows_kernel<<<Bsz, 256, 0, stream>>>(
        y_f, gstate, nullptr, mha_ln_w, mha_ln_b, x1_f, x1_bf, Dd);
    gemm_bf16_kernel<<<dim3(32, 32), 256, 0, stream>>>(
        x1_bf, Dd, sffw1_bf, Dd, 1, t_f, DInn, nullptr, Bsz, DInn, Dd);
    relu_bias_kernel<<<(Bsz * DInn / 4) / 256, 256, 0, stream>>>(
        t_f, s_ff_b1, h1_bf, Bsz * DInn, DInn);
    gemm_bf16_kernel<<<dim3(32, 4), 256, 0, stream>>>(
        h1_bf, DInn, sffw2_bf, DInn, 1, u_f, Dd, nullptr, Bsz, Dd, DInn);
    ln_rows_kernel<<<Bsz, 256, 0, stream>>>(
        u_f, x1_f, s_ff_b2, s_ff_lnw, s_ff_lnb, nullptr, x2_bf, Dd);
    gemm_bf16_kernel<<<dim3(32, (NSv + 63) / 64), 256, 0, stream>>>(
        x2_bf, Dd, sproj_bf, Dd, 1, lgt_f, NSv, nullptr, Bsz, NSv, Dd);
    ln_rows_kernel<<<Bsz, 256, 0, stream>>>(
        lgt_f, nullptr, nullptr, s_out_w, s_out_b, out_s, nullptr, NSv);

    // ---- operator stream (reuses chain buffers) ----
    gemm_bf16_kernel<<<dim3(32, 4), 256, 0, stream>>>(
        aggo_bf, 2048, wcomb_bf, 256, 0, y_f, Dd, nullptr, Bsz, Dd, 2048);
    ln_rows_kernel<<<Bsz, 256, 0, stream>>>(
        y_f, gstate, nullptr, mha_ln_w, mha_ln_b, x1_f, x1_bf, Dd);
    gemm_bf16_kernel<<<dim3(32, 32), 256, 0, stream>>>(
        x1_bf, Dd, offw1_bf, Dd, 1, t_f, DInn, nullptr, Bsz, DInn, Dd);
    relu_bias_kernel<<<(Bsz * DInn / 4) / 256, 256, 0, stream>>>(
        t_f, o_ff_b1, h1_bf, Bsz * DInn, DInn);
    gemm_bf16_kernel<<<dim3(32, 4), 256, 0, stream>>>(
        h1_bf, DInn, offw2_bf, DInn, 1, u_f, Dd, nullptr, Bsz, Dd, DInn);
    ln_rows_kernel<<<Bsz, 256, 0, stream>>>(
        u_f, x1_f, o_ff_b2, o_ff_lnw, o_ff_lnb, nullptr, x2_bf, Dd);
    gemm_bf16_kernel<<<dim3(32, 1), 256, 0, stream>>>(
        x2_bf, Dd, oproj_bf, Dd, 1, lgt_f, NOv, nullptr, Bsz, NOv, Dd);
    ln_rows_kernel<<<Bsz, 256, 0, stream>>>(
        lgt_f, nullptr, nullptr, o_out_w, o_out_b, out_o, nullptr, NOv);
}
